// GlobalAttentionLayer_41205916237953
// MI455X (gfx1250) — matrix operand order repaired and checked by reading, not compiled, not run on hardware
//
#include <hip/hip_runtime.h>
#include <stdint.h>

typedef __attribute__((ext_vector_type(16))) _Float16 v16h;
typedef __attribute__((ext_vector_type(8)))  _Float16 v8h;
typedef __attribute__((ext_vector_type(8)))  float    v8f;

#define HDIM 256
#define NSEQ 1024
#define BATCH 32

__device__ __forceinline__ v16h cat8(v8h a, v8h b) {
  return __builtin_shufflevector(a, b, 0, 1, 2, 3, 4, 5, 6, 7, 8, 9, 10, 11, 12, 13, 14, 15);
}

// D = A*B + C with f16 hi/lo split emulation of fp32 (drop lo*lo term).
__device__ __forceinline__ v8f wmma3(v16h ah, v16h al, v16h bh, v16h bl, v8f c) {
  c = __builtin_amdgcn_wmma_f32_16x16x32_f16(false, ah, false, bh, (short)0, c, false, false);
  c = __builtin_amdgcn_wmma_f32_16x16x32_f16(false, ah, false, bl, (short)0, c, false, false);
  c = __builtin_amdgcn_wmma_f32_16x16x32_f16(false, al, false, bh, (short)0, c, false, false);
  return c;
}

__device__ __forceinline__ float wave_sum(float v) {
  #pragma unroll
  for (int off = 16; off > 0; off >>= 1) v += __shfl_xor(v, off);
  return v;
}
__device__ __forceinline__ float wave_max(float v) {
  #pragma unroll
  for (int off = 16; off > 0; off >>= 1) v = fmaxf(v, __shfl_xor(v, off));
  return v;
}

// ---------------- kernel 0a: fp32 -> (hi,lo) f16 split ----------------
__global__ __launch_bounds__(256) void split_kernel(const float* __restrict__ x,
                                                    _Float16* __restrict__ hi,
                                                    _Float16* __restrict__ lo, int n) {
  int i = blockIdx.x * blockDim.x + threadIdx.x;
  if (i < n) {
    float v = x[i];
    _Float16 h = (_Float16)v;
    hi[i] = h;
    lo[i] = (_Float16)(v - (float)h);
  }
}

// ---------------- kernel 0b: weight split + transpose (W[d][h] -> Wt[h][d]) ----------------
__global__ __launch_bounds__(256) void split_w_kernel(const float* __restrict__ w,
                                                      _Float16* __restrict__ hiT,
                                                      _Float16* __restrict__ loT) {
  int i = blockIdx.x * 256 + threadIdx.x;  // over H*H
  int d = i >> 8, h = i & 255;
  float v = w[i];
  _Float16 hh = (_Float16)v;
  hiT[h * HDIM + d] = hh;
  loT[h * HDIM + d] = (_Float16)(v - (float)hh);
}

// ---------------- kernel 1: fused QKV projection (split-f16 WMMA) ----------------
struct QkvArgs {
  const _Float16* Fhi; const _Float16* Flo;
  const _Float16* WhT[3]; const _Float16* WlT[3];
  const float* bias[3];
  _Float16* Qh; _Float16* Ql;
  _Float16* Kh; _Float16* Kl;
  _Float16* VhT; _Float16* VlT;  // [B][H][N] transposed per batch
};

__global__ __launch_bounds__(256) void qkv_kernel(QkvArgs a) {
  const int lane = threadIdx.x & 31;
  const int wave = threadIdx.x >> 5;
  const int g = lane >> 4, l16 = lane & 15;
  const int rowBase = blockIdx.x * 32;

  // 24 groups of 4 N-tiles: grp -> (mat, mt, ntq); A-fragment shared across 4 tiles
  for (int grp = wave; grp < 24; grp += 8) {
    const int mat = grp >> 3;
    const int rem = grp & 7;
    const int mt  = rem >> 2;
    const int ntq = rem & 3;
    const int arow = rowBase + mt * 16 + l16;
    v8f acc[4] = {};
    for (int kc = 0; kc < 8; ++kc) {
      const int k0 = kc * 32;
      const _Float16* pa_h = a.Fhi + (size_t)arow * HDIM + k0 + g * 8;
      const _Float16* pa_l = a.Flo + (size_t)arow * HDIM + k0 + g * 8;
      v16h ah = cat8(*(const v8h*)pa_h, *(const v8h*)(pa_h + 16));
      v16h al = cat8(*(const v8h*)pa_l, *(const v8h*)(pa_l + 16));
      #pragma unroll
      for (int j = 0; j < 4; ++j) {
        const int n0 = (ntq * 4 + j) * 16;
        const size_t widx = (size_t)(n0 + l16) * HDIM + k0 + g * 8;  // Wt row-major, table k order
        v16h bh = cat8(*(const v8h*)(a.WhT[mat] + widx), *(const v8h*)(a.WhT[mat] + widx + 16));
        v16h bl = cat8(*(const v8h*)(a.WlT[mat] + widx), *(const v8h*)(a.WlT[mat] + widx + 16));
        acc[j] = wmma3(ah, al, bh, bl, acc[j]);
      }
    }
    #pragma unroll
    for (int j = 0; j < 4; ++j) {
      const int col = (ntq * 4 + j) * 16 + l16;
      const float bval = a.bias[mat][col];
      if (mat == 2) {
        // V: transposed store; the 8 D-rows per lane are 8 consecutive keys -> v8h store
        v8h hv, lv;
        #pragma unroll
        for (int i = 0; i < 8; ++i) {
          float v = acc[j][i] + bval;
          _Float16 h = (_Float16)v;
          hv[i] = h;
          lv[i] = (_Float16)(v - (float)h);
        }
        const int row0 = rowBase + mt * 16 + g * 8;  // first of 8 consecutive rows
        const int bb = row0 >> 10;
        const int n  = row0 & 1023;
        const size_t tidx = ((size_t)bb * HDIM + col) * NSEQ + n;
        *(v8h*)(a.VhT + tidx) = hv;
        *(v8h*)(a.VlT + tidx) = lv;
      } else {
        _Float16* Oh = (mat == 0) ? a.Qh : a.Kh;
        _Float16* Ol = (mat == 0) ? a.Ql : a.Kl;
        const float scale = (mat == 0) ? 0.0625f : 1.0f;  // fold 1/sqrt(256) into Q
        #pragma unroll
        for (int i = 0; i < 8; ++i) {
          const int row = rowBase + mt * 16 + g * 8 + i;
          float v = (acc[j][i] + bval) * scale;
          _Float16 h = (_Float16)v;
          Oh[(size_t)row * HDIM + col] = h;
          Ol[(size_t)row * HDIM + col] = (_Float16)(v - (float)h);
        }
      }
    }
  }
}

// ---------------- kernel 2: attention (scores + masked softmax + PV) ----------------
__global__ __launch_bounds__(256) void attn_kernel(
    const _Float16* __restrict__ Qhi, const _Float16* __restrict__ Qlo,
    const _Float16* __restrict__ Khi, const _Float16* __restrict__ Klo,
    const _Float16* __restrict__ VhT, const _Float16* __restrict__ VlT,
    const int* __restrict__ masks, float* __restrict__ out) {
  extern __shared__ char smem_raw[];
  float* S   = (float*)smem_raw;                    // [32][1024] scores; later row r = [hi x1024 | lo x1024] f16
  int* mrowI = (int*)(smem_raw + 32 * NSEQ * 4);    // [1024] key masks

  const int lane = threadIdx.x & 31;
  const int wave = threadIdx.x >> 5;
  const int g = lane >> 4, l16 = lane & 15;
  const int b  = blockIdx.y;
  const int q0 = blockIdx.x * 32;
  const size_t batchOff = (size_t)b * NSEQ * HDIM;

  // Stage the key-mask row via the CDNA5 async global->LDS path (ASYNCcnt-tracked).
  {
    const int* gp = masks + b * NSEQ;
    #pragma unroll
    for (int j = 0; j < 4; ++j) {
      uint32_t loff  = (uint32_t)(uintptr_t)(mrowI + threadIdx.x + j * 256);
      uint64_t gaddr = (uint64_t)(uintptr_t)(gp + threadIdx.x + j * 256);
      asm volatile("global_load_async_to_lds_b32 %0, %1, off" :: "v"(loff), "v"(gaddr) : "memory");
    }
    asm volatile("s_wait_asynccnt 0" ::: "memory");
  }
  __syncthreads();

  const int mt = wave & 1;

  // Phase A: S[32][1024] = Qs * K^T (scale folded into Q); 8 accumulators per wave
  {
    const int ntBase = (wave >> 1) * 16;
    const int arow = q0 + mt * 16 + l16;
    for (int sub = 0; sub < 2; ++sub) {
      v8f acc[8] = {};
      for (int kc = 0; kc < 8; ++kc) {
        const int k0 = kc * 32;
        const _Float16* pa_h = Qhi + batchOff + (size_t)arow * HDIM + k0 + g * 8;
        const _Float16* pa_l = Qlo + batchOff + (size_t)arow * HDIM + k0 + g * 8;
        v16h ah = cat8(*(const v8h*)pa_h, *(const v8h*)(pa_h + 16));
        v16h al = cat8(*(const v8h*)pa_l, *(const v8h*)(pa_l + 16));
        #pragma unroll
        for (int j = 0; j < 8; ++j) {
          const int krow = (ntBase + sub * 8 + j) * 16 + l16;
          v16h bh = cat8(*(const v8h*)(Khi + batchOff + (size_t)krow * HDIM + k0 + g * 8), *(const v8h*)(Khi + batchOff + (size_t)krow * HDIM + k0 + 16 + g * 8));
          v16h bl = cat8(*(const v8h*)(Klo + batchOff + (size_t)krow * HDIM + k0 + g * 8), *(const v8h*)(Klo + batchOff + (size_t)krow * HDIM + k0 + 16 + g * 8));
          acc[j] = wmma3(ah, al, bh, bl, acc[j]);
        }
      }
      #pragma unroll
      for (int j = 0; j < 8; ++j) {
        const int m0 = (ntBase + sub * 8 + j) * 16;
        #pragma unroll
        for (int i = 0; i < 8; ++i)
          S[(size_t)(mt * 16 + g * 8 + i) * NSEQ + m0 + l16] = acc[j][i];
      }
    }
  }
  __syncthreads();

  // Phase B: masked softmax per query row; repack row in place as f16 [hi|lo]
  for (int rr = 0; rr < 4; ++rr) {
    const int r = wave * 4 + rr;
    const float mq = (float)mrowI[q0 + r];
    float* Sr = S + (size_t)r * NSEQ;
    float vals[32];
    float mx = -3.0e38f;
    #pragma unroll
    for (int j = 0; j < 32; ++j) {
      const int m = lane + j * 32;
      const float am = mq * (float)mrowI[m];
      // exact reference additive: am - 1e5*(1-am) = am*100001 - 100000
      const float v = Sr[m] + am * 100001.0f - 100000.0f;
      vals[j] = v;
      mx = fmaxf(mx, v);
    }
    mx = wave_max(mx);
    float sum = 0.f;
    #pragma unroll
    for (int j = 0; j < 32; ++j) { vals[j] = __expf(vals[j] - mx); sum += vals[j]; }
    sum = wave_sum(sum);
    const float inv = 1.0f / sum;
    _Float16* Pr = (_Float16*)Sr;  // row region reuse: [0..1023]=hi, [1024..2047]=lo
    #pragma unroll
    for (int j = 0; j < 32; ++j) {
      const int m = lane + j * 32;
      const float p = vals[j] * inv;
      _Float16 h = (_Float16)p;
      Pr[m] = h;
      Pr[NSEQ + m] = (_Float16)(p - (float)h);
    }
  }
  __syncthreads();

  // Phase C: out[32][256] = P * V ; A-fragments from LDS (vector loads), B from V^T (contiguous)
  {
    const int ntB = (wave >> 1) * 4;
    const _Float16* Ph = (const _Float16*)(S + (size_t)(mt * 16 + l16) * NSEQ);
    v8f acc[4] = {};
    for (int kc = 0; kc < 32; ++kc) {
      const int k0 = kc * 32;
      v16h ah = cat8(*(const v8h*)(Ph + k0 + g * 8), *(const v8h*)(Ph + k0 + 16 + g * 8));
      v16h al = cat8(*(const v8h*)(Ph + NSEQ + k0 + g * 8), *(const v8h*)(Ph + NSEQ + k0 + 16 + g * 8));
      #pragma unroll
      for (int j = 0; j < 4; ++j) {
        const int col = (ntB + j) * 16 + l16;
        const size_t vidx = ((size_t)b * HDIM + col) * NSEQ + k0 + g * 8;
        v16h bh = cat8(*(const v8h*)(VhT + vidx), *(const v8h*)(VhT + vidx + 16));
        v16h bl = cat8(*(const v8h*)(VlT + vidx), *(const v8h*)(VlT + vidx + 16));
        acc[j] = wmma3(ah, al, bh, bl, acc[j]);
      }
    }
    #pragma unroll
    for (int j = 0; j < 4; ++j) {
      const int col = (ntB + j) * 16 + l16;
      #pragma unroll
      for (int i = 0; i < 8; ++i) {
        const int r = q0 + mt * 16 + g * 8 + i;
        out[((size_t)b * NSEQ + r) * HDIM + col] = acc[j][i];
      }
    }
  }
}

// ---------------- kernel 3: in-place LayerNorm + ReLU (one wave per row) ----------------
__global__ __launch_bounds__(256) void ln_relu_kernel(float* __restrict__ out,
                                                      const float* __restrict__ gamma,
                                                      const float* __restrict__ beta) {
  const int lane = threadIdx.x & 31;
  const int wave = threadIdx.x >> 5;
  const size_t row = (size_t)blockIdx.x * 8 + wave;
  float* p = out + row * HDIM;
  float x[8];
  float s = 0.f;
  #pragma unroll
  for (int j = 0; j < 8; ++j) { x[j] = p[lane + j * 32]; s += x[j]; }
  s = wave_sum(s);
  const float mu = s * (1.0f / HDIM);
  float v = 0.f;
  #pragma unroll
  for (int j = 0; j < 8; ++j) { const float d = x[j] - mu; v += d * d; }
  v = wave_sum(v);
  const float rs = rsqrtf(v * (1.0f / HDIM) + 1e-5f);
  #pragma unroll
  for (int j = 0; j < 8; ++j) {
    const int c = lane + j * 32;
    const float y = (x[j] - mu) * rs * gamma[c] + beta[c];
    p[c] = fmaxf(y, 0.f);
  }
}

extern "C" void kernel_launch(void* const* d_in, const int* in_sizes, int n_in,
                              void* d_out, int out_size, void* d_ws, size_t ws_size,
                              hipStream_t stream) {
  const float* feats = (const float*)d_in[0];
  const int*   masks = (const int*)d_in[1];
  const float* Wq = (const float*)d_in[2];
  const float* bq = (const float*)d_in[3];
  const float* Wk = (const float*)d_in[4];
  const float* bk = (const float*)d_in[5];
  const float* Wv = (const float*)d_in[6];
  const float* bv = (const float*)d_in[7];
  const float* gamma = (const float*)d_in[8];
  const float* beta  = (const float*)d_in[9];
  float* out = (float*)d_out;

  const size_t NR = (size_t)BATCH * NSEQ;  // 32768 rows
  const size_t FE = NR * HDIM;             // 8,388,608 elements
  const size_t WE = (size_t)HDIM * HDIM;   // 65,536 elements

  char* ws = (char*)d_ws;
  size_t off = 0;
  auto alloc16 = [&](size_t elems) -> _Float16* {
    _Float16* p = (_Float16*)(ws + off);
    off += ((elems * 2 + 255) / 256) * 256;
    return p;
  };
  _Float16* Fhi = alloc16(FE);
  _Float16* Flo = alloc16(FE);
  const float* Wsrc[3] = {Wq, Wk, Wv};
  const float* bsrc[3] = {bq, bk, bv};
  _Float16 *WhT[3], *WlT[3];
  for (int m = 0; m < 3; ++m) { WhT[m] = alloc16(WE); WlT[m] = alloc16(WE); }
  _Float16* Qh  = alloc16(FE); _Float16* Ql  = alloc16(FE);
  _Float16* Kh  = alloc16(FE); _Float16* Kl  = alloc16(FE);
  _Float16* VhT = alloc16(FE); _Float16* VlT = alloc16(FE);

  split_kernel<<<(int)((FE + 255) / 256), 256, 0, stream>>>(feats, Fhi, Flo, (int)FE);
  for (int m = 0; m < 3; ++m)
    split_w_kernel<<<(int)(WE / 256), 256, 0, stream>>>(Wsrc[m], WhT[m], WlT[m]);

  QkvArgs qa;
  qa.Fhi = Fhi; qa.Flo = Flo;
  for (int m = 0; m < 3; ++m) { qa.WhT[m] = WhT[m]; qa.WlT[m] = WlT[m]; qa.bias[m] = bsrc[m]; }
  qa.Qh = Qh; qa.Ql = Ql; qa.Kh = Kh; qa.Kl = Kl; qa.VhT = VhT; qa.VlT = VlT;
  qkv_kernel<<<(int)(NR / 32), 256, 0, stream>>>(qa);

  const size_t lds = 32 * NSEQ * sizeof(float) + NSEQ * sizeof(int);  // 132 KB
  attn_kernel<<<dim3(NSEQ / 32, BATCH), 256, lds, stream>>>(
      Qh, Ql, Kh, Kl, VhT, VlT, masks, out);

  ln_relu_kernel<<<(int)(NR / 8), 256, 0, stream>>>(out, gamma, beta);
}
